// SelfAttentionLayer_223338300097
// MI455X (gfx1250) — matrix operand order repaired and checked by reading, not compiled, not run on hardware
//
#include <hip/hip_runtime.h>

// ---------------------------------------------------------------------------
// Self-attention (B=8, S=2048, D=1024), fp32 in/out, bf16 WMMA compute.
// MI455X (gfx1250): wave32, v_wmma_f32_16x16x32_bf16, TDM K-block streaming,
// 192MB L2 holds the whole KV set so V fragments read straight from global.
//
// Roofline: ~240 GFLOP total (103 QKV + 69 scores + 69 context) vs ~hundreds
// of MB of traffic at 23.3 TB/s -> compute bound -> everything through WMMA.
// ---------------------------------------------------------------------------

constexpr int B_ = 8;
constexpr int S_ = 2048;
constexpr int D_ = 1024;

typedef __attribute__((ext_vector_type(4)))  __bf16 v4bf;
typedef __attribute__((ext_vector_type(8)))  __bf16 v8bf;
typedef __attribute__((ext_vector_type(16))) __bf16 v16bf;
typedef __attribute__((ext_vector_type(8)))  float  v8f;

typedef __attribute__((ext_vector_type(4))) unsigned int u32x4;
typedef __attribute__((ext_vector_type(8))) int          i32x8;
typedef __attribute__((ext_vector_type(4))) int          i32x4;

#if defined(__HIP_DEVICE_COMPILE__) && __has_builtin(__builtin_amdgcn_tensor_load_to_lds)
#define HAS_TDM 1
#else
#define HAS_TDM 0
#endif

// LDS layout for attn kernel (dynamic): K double buffer with TDM row padding.
// Row = 2048B data + 2x8B pad (pad_interval=256dw, pad_amount=2dw) = 2064B.
constexpr unsigned KROW_B   = 2064;                 // padded key-row stride
constexpr unsigned KBUF_B   = 32 * KROW_B;          // 66048 B per buffer
constexpr unsigned SS_OFF   = 2 * KBUF_B;           // 132096: float sS[512]
constexpr unsigned SP_OFF   = SS_OFF + 512 * 4;     // 134144: bf16 sP[512]
constexpr unsigned SA_OFF   = SP_OFF + 512 * 2;     // 135168: float sAlpha[16]
constexpr unsigned SL_OFF   = SA_OFF + 16 * 4;      // 135232: float sLinv[16]
constexpr unsigned SMEM_B   = SL_OFF + 16 * 4;      // 135296 total

// A-fragment (16x32 bf16): lane row = lane%16; K chunk [c0,c0+16)
// with c0=(lane/16)*16 -> two contiguous 16-byte loads.
static __device__ __forceinline__ v16bf load_a_frag(const __bf16* p) {
  v16bf r;
  ((v8bf*)&r)[0] = *(const v8bf*)(p);
  ((v8bf*)&r)[1] = *(const v8bf*)(p + 8);
  return r;
}

#if HAS_TDM
// Issue one TDM 2D tile load: 32 key rows x 1024 bf16, row stride D_, into
// LDS at lds_off_bytes with 2-dword pad every 256 dwords (bank decorrelation).
static __device__ __forceinline__ void tdm_issue_k(const __bf16* gptr,
                                                   unsigned lds_off_bytes) {
  const unsigned long long ga = (unsigned long long)(uintptr_t)gptr;
  u32x4 g0;
  g0[0] = 1u;                                            // count=1 (valid D#)
  g0[1] = lds_off_bytes;                                 // lds_addr
  g0[2] = (unsigned)(ga & 0xffffffffu);                  // global_addr[31:0]
  g0[3] = (unsigned)((ga >> 32) & 0x1ffffffu) | (2u << 30);  // ga hi | type=2
  i32x8 g1;
  g1[0] = (int)((1u << 16) |            // data_size=1 (2 bytes)
                (1u << 20) |            // pad_enable
                (7u << 22) |            // pad_interval: 7 -> 256 dwords
                (1u << 25));            // pad_amount:   1 -> 2 dwords
  g1[1] = (int)(1024u << 16);                    // tensor_dim0[15:0] @ bit48
  g1[2] = (int)(32u << 16);                      // dim0 hi | tensor_dim1 lo
  g1[3] = (int)(1024u << 16);                    // dim1 hi | tile_dim0=1024
  g1[4] = 32;                                    // tile_dim1=32, tile_dim2=0
  g1[5] = 1024;                                  // tensor_dim0_stride lo32
  g1[6] = 0;                                     // stride0 hi | stride1 lo
  g1[7] = 0;                                     // stride1 hi
  i32x4 gz = {0, 0, 0, 0};                       // groups 2/3: 2D tile
#if __clang_major__ >= 23
  i32x8 gz8 = {0, 0, 0, 0, 0, 0, 0, 0};
  __builtin_amdgcn_tensor_load_to_lds(g0, g1, gz, gz, gz8, 0);
#else
  __builtin_amdgcn_tensor_load_to_lds(g0, g1, gz, gz, 0);
#endif
}
#endif

// ---------------------------------------------------------------------------
// Kernel 1: x fp32 -> bf16 (vectorized)
// ---------------------------------------------------------------------------
__global__ __launch_bounds__(256)
void cvt_x_kernel(const float4* __restrict__ in, __bf16* __restrict__ out, int n4) {
  const int i = blockIdx.x * 256 + threadIdx.x;
  if (i >= n4) return;
  const float4 v = in[i];
  v4bf o = { (__bf16)v.x, (__bf16)v.y, (__bf16)v.z, (__bf16)v.w };
  *(v4bf*)(out + (size_t)4 * i) = o;
}

// ---------------------------------------------------------------------------
// Kernel 2: W[k][n] fp32 -> Wt[n][k] bf16 (LDS-tiled transpose)
// grid (32,32,3), block (32,8)
// ---------------------------------------------------------------------------
__global__ __launch_bounds__(256)
void cvt_w_kernel(const float* __restrict__ Wq, const float* __restrict__ Wk,
                  const float* __restrict__ Wv, __bf16* __restrict__ WtOut) {
  __shared__ float tile[32][33];
  const float* W = (blockIdx.z == 0) ? Wq : (blockIdx.z == 1) ? Wk : Wv;
  __bf16* O = WtOut + (size_t)blockIdx.z * D_ * D_;
  const int bx = blockIdx.x * 32;
  const int by = blockIdx.y * 32;
  const int tx = threadIdx.x, ty = threadIdx.y;
#pragma unroll
  for (int r = ty; r < 32; r += 8)
    tile[r][tx] = W[(size_t)(by + r) * D_ + bx + tx];
  __syncthreads();
#pragma unroll
  for (int r = ty; r < 32; r += 8)
    O[(size_t)(bx + r) * D_ + by + tx] = (__bf16)tile[tx][r];
}

// ---------------------------------------------------------------------------
// Kernel 3: QKV projection. Y = X(bf16) @ W + b via WMMA bf16->f32.
//   z=0 -> Q row-major, z=1 -> K row-major, z=2 -> V transposed [b][d][s]
// ---------------------------------------------------------------------------
__global__ __launch_bounds__(256)
void proj_kernel(const __bf16* __restrict__ Xb, const __bf16* __restrict__ WtAll,
                 const float* __restrict__ bq, const float* __restrict__ bk,
                 const float* __restrict__ bv,
                 __bf16* __restrict__ Qb, __bf16* __restrict__ Kb,
                 __bf16* __restrict__ Vt) {
  const int which = blockIdx.z;
  const __bf16* W = WtAll + (size_t)which * D_ * D_;
  const float* bias = (which == 0) ? bq : (which == 1) ? bk : bv;

  const int tid  = threadIdx.x;
  const int wv   = tid >> 5;
  const int lane = tid & 31;
  const int hl   = lane >> 4;
  const int l16  = lane & 15;

  const int m0 = blockIdx.x * 128 + wv * 16;
  const int n0 = blockIdx.y * 64;

  v8f acc[4];
#pragma unroll
  for (int t = 0; t < 4; ++t)
#pragma unroll
    for (int j = 0; j < 8; ++j) acc[t][j] = 0.f;

  const __bf16* arow = Xb + (size_t)(m0 + l16) * D_;
  for (int kc = 0; kc < D_; kc += 32) {
    const v16bf af = load_a_frag(arow + kc + hl * 16);
#pragma unroll
    for (int t = 0; t < 4; ++t) {
      const __bf16* bp = W + (size_t)(n0 + t * 16 + l16) * D_ + kc + hl * 16;
      const v16bf bfrag = *(const v16bf*)bp;
      acc[t] = __builtin_amdgcn_wmma_f32_16x16x32_bf16(
          false, af, false, bfrag, (short)0, acc[t], false, false);
    }
  }

#pragma unroll
  for (int t = 0; t < 4; ++t) {
    const float bsv = bias[n0 + t * 16 + l16];
#pragma unroll
    for (int j = 0; j < 8; ++j) acc[t][j] += bsv;
  }

  if (which < 2) {
    __bf16* O = (which == 0) ? Qb : Kb;
#pragma unroll
    for (int t = 0; t < 4; ++t) {
      const int col = n0 + t * 16 + l16;
#pragma unroll
      for (int j = 0; j < 8; ++j) {
        const int row = m0 + j + hl * 8;
        O[(size_t)row * D_ + col] = (__bf16)acc[t][j];
      }
    }
  } else {
#pragma unroll
    for (int t = 0; t < 4; ++t) {
      const int col = n0 + t * 16 + l16;     // d index
#pragma unroll
      for (int j = 0; j < 8; ++j) {
        const int row = m0 + j + hl * 8;     // flattened b*S + s
        const int bb = row >> 11;
        const int ss = row & 2047;
        Vt[((size_t)bb * D_ + col) * S_ + ss] = (__bf16)acc[t][j];
      }
    }
  }
}

// ---------------------------------------------------------------------------
// Kernel 4: flash attention with TDM-streamed K blocks.
// grid (S/16, B), block 256 (8 waves). Wave w owns output d-slice
// [w*128, w*128+128). K blocks (32 keys x 1024 d, bf16) are DMA'd into a
// double-buffered padded LDS tile by wave 0 via tensor_load_to_lds and
// synchronized with s_wait_tensorcnt; V fragments stream from global (L2).
// ---------------------------------------------------------------------------
__global__ __launch_bounds__(256)
void attn_kernel(const __bf16* __restrict__ Qb, const __bf16* __restrict__ Kb,
                 const __bf16* __restrict__ Vt, float* __restrict__ Out) {
  extern __shared__ char smem[];
  float*  sS     = (float*)(smem + SS_OFF);
  __bf16* sP     = (__bf16*)(smem + SP_OFF);
  float*  sAlpha = (float*)(smem + SA_OFF);
  float*  sLinv  = (float*)(smem + SL_OFF);

  const int tid  = threadIdx.x;
  const int wv   = tid >> 5;
  const int lane = tid & 31;
  const int hl   = lane >> 4;
  const int l16  = lane & 15;
  const int b    = blockIdx.y;
  const int q0   = blockIdx.x * 16;
  constexpr int NKB = S_ / 32;

  const __bf16* Kbase = Kb + (size_t)b * S_ * D_;
  const __bf16* Vbase = Vt + (size_t)b * D_ * S_;   // [d][s]

#if HAS_TDM
  const unsigned lds_base = (unsigned)(uintptr_t)smem;
  if (wv == 0) tdm_issue_k(Kbase, lds_base);        // prologue: block 0 -> buf 0
#endif

  // Preload Q A-fragments for this wave's 128-wide d slice (reused every step)
  v16bf aQ[4];
  {
    const __bf16* qrow = Qb + ((size_t)b * S_ + q0 + l16) * D_ + wv * 128 + hl * 16;
#pragma unroll
    for (int c = 0; c < 4; ++c) aQ[c] = load_a_frag(qrow + c * 32);
  }

  v8f acc[8];
#pragma unroll
  for (int t = 0; t < 8; ++t)
#pragma unroll
    for (int j = 0; j < 8; ++j) acc[t][j] = 0.f;

  float m_run = -__builtin_inff();   // live in lanes tid<16 only
  float l_run = 0.f;

  for (int kb = 0; kb < NKB; ++kb) {
    const int cur = kb & 1;
    // --- zero score tile ---
    sS[tid] = 0.f;
    sS[tid + 256] = 0.f;

#if HAS_TDM
    if (wv == 0) {
      if (kb + 1 < NKB) {
        // Safe: buffer cur^1 was last read before the post-score barrier of
        // iteration kb-1, which this wave has already passed.
        tdm_issue_k(Kbase + (size_t)(kb + 1) * 32 * D_, lds_base + (cur ^ 1) * KBUF_B);
        __builtin_amdgcn_s_wait_tensorcnt((short)1);   // current buffer ready
      } else {
        __builtin_amdgcn_s_wait_tensorcnt((short)0);
      }
    }
#endif
    __syncthreads();

    if (kb + 1 < NKB)  // prefetch next V block (global_prefetch_b8)
      __builtin_prefetch(Vbase + (size_t)(wv * 128 + l16) * S_ + (kb + 1) * 32, 0, 0);

    // --- partial scores over this wave's d-slice: 2 key n-tiles x 4 k-chunks ---
#if HAS_TDM
    const char* kbuf = smem + cur * KBUF_B;
#endif
#pragma unroll
    for (int t = 0; t < 2; ++t) {
      v8f cfrag;
#pragma unroll
      for (int j = 0; j < 8; ++j) cfrag[j] = 0.f;
      const int key = t * 16 + l16;   // key row within block
#pragma unroll
      for (int ch = 0; ch < 4; ++ch) {
        const unsigned dby = (unsigned)(wv * 256 + ch * 64 + hl * 32);  // d bytes
#if HAS_TDM
        const v16bf bfrag = *(const v16bf*)(kbuf + (unsigned)key * KROW_B +
                                            dby + ((dby >> 10) << 3));
#else
        const v16bf bfrag = *(const v16bf*)((const char*)Kbase +
            ((size_t)(kb * 32 + key) * D_) * 2 + dby);
#endif
        cfrag = __builtin_amdgcn_wmma_f32_16x16x32_bf16(
            false, aQ[ch], false, bfrag, (short)0, cfrag, false, false);
      }
#pragma unroll
      for (int j = 0; j < 8; ++j)
        atomicAdd(&sS[(j + hl * 8) * 32 + t * 16 + l16], cfrag[j]);  // ds_add_f32
    }
    __syncthreads();

    // --- online softmax, one row per lane (lanes 0..15 of wave 0) ---
    if (tid < 16) {
      const int r = tid;
      float mx = m_run;
      float sc[32];
#pragma unroll
      for (int n = 0; n < 32; ++n) { sc[n] = sS[r * 32 + n]; mx = fmaxf(mx, sc[n]); }
      const float alpha = __expf(m_run - mx);
      float sum = 0.f;
#pragma unroll
      for (int n = 0; n < 32; ++n) {
        const float p = __expf(sc[n] - mx);
        sum += p;
        sP[r * 32 + n] = (__bf16)p;
      }
      l_run = l_run * alpha + sum;
      m_run = mx;
      sAlpha[r] = alpha;
      if (kb == NKB - 1) sLinv[r] = 1.f / l_run;
    }
    __syncthreads();

    // --- rescale accumulators by alpha[row] ---
    float av[8];
#pragma unroll
    for (int j = 0; j < 8; ++j) av[j] = sAlpha[hl * 8 + j];
#pragma unroll
    for (int t = 0; t < 8; ++t)
#pragma unroll
      for (int j = 0; j < 8; ++j) acc[t][j] *= av[j];

    // --- context: acc[t] += P(16x32) x V(32x16) per d-tile ---
    const v16bf pf = load_a_frag(sP + l16 * 32 + hl * 16);
#pragma unroll
    for (int t = 0; t < 8; ++t) {
      const int d = wv * 128 + t * 16 + l16;
      const v16bf vfrag =
          *(const v16bf*)(Vbase + (size_t)d * S_ + kb * 32 + hl * 16);
      acc[t] = __builtin_amdgcn_wmma_f32_16x16x32_bf16(
          false, pf, false, vfrag, (short)0, acc[t], false, false);
    }
    // next-iter zeroing of sS / TDM overwrite are ordered by the barriers above
  }

  // --- epilogue: out = acc / l ---
  float linv[8];
#pragma unroll
  for (int j = 0; j < 8; ++j) linv[j] = sLinv[hl * 8 + j];
  float* orow = Out + ((size_t)b * S_ + q0) * D_;
#pragma unroll
  for (int t = 0; t < 8; ++t) {
    const int col = wv * 128 + t * 16 + l16;
#pragma unroll
    for (int j = 0; j < 8; ++j)
      orow[(size_t)(j + hl * 8) * D_ + col] = acc[t][j] * linv[j];
  }
}

// ---------------------------------------------------------------------------
// Launch
// ---------------------------------------------------------------------------
extern "C" void kernel_launch(void* const* d_in, const int* in_sizes, int n_in,
                              void* d_out, int out_size, void* d_ws, size_t ws_size,
                              hipStream_t stream) {
  const float* x  = (const float*)d_in[0];
  const float* Wq = (const float*)d_in[1];
  const float* bq = (const float*)d_in[2];
  const float* Wk = (const float*)d_in[3];
  const float* bk = (const float*)d_in[4];
  const float* Wv = (const float*)d_in[5];
  const float* bv = (const float*)d_in[6];
  float* out = (float*)d_out;

  const size_t Nx = (size_t)B_ * S_ * D_;          // 16,777,216
  char* p = (char*)d_ws;
  __bf16* Xb  = (__bf16*)p;  p += Nx * sizeof(__bf16);                  // 32 MB
  __bf16* Wt  = (__bf16*)p;  p += (size_t)3 * D_ * D_ * sizeof(__bf16); //  6 MB
  __bf16* Qb  = (__bf16*)p;  p += Nx * sizeof(__bf16);                  // 32 MB
  __bf16* Kbf = (__bf16*)p;  p += Nx * sizeof(__bf16);                  // 32 MB
  __bf16* Vtb = (__bf16*)p;  p += Nx * sizeof(__bf16);                  // 32 MB
  (void)ws_size; (void)in_sizes; (void)n_in; (void)out_size;

  cvt_x_kernel<<<(int)(Nx / 4 / 256), 256, 0, stream>>>(
      (const float4*)x, Xb, (int)(Nx / 4));
  cvt_w_kernel<<<dim3(32, 32, 3), dim3(32, 8), 0, stream>>>(Wq, Wk, Wv, Wt);
  proj_kernel<<<dim3((B_ * S_) / 128, D_ / 64, 3), 256, 0, stream>>>(
      Xb, Wt, bq, bk, bv, Qb, Kbf, Vtb);
  attn_kernel<<<dim3(S_ / 16, B_), 256, SMEM_B, stream>>>(Qb, Kbf, Vtb, out);
}
